// RGCN_30520037605680
// MI455X (gfx1250) — compile-verified, matrix operand order checked by reading, not run on hardware
//
#include <hip/hip_runtime.h>

// RGCN (2-layer, R=8, mean aggregation) for MI455X / gfx1250.
//
// Pipeline (transform-first — keeps layer-2 edge traffic at 128ch, and T1
// (184MB) ~fits the 192MB L2 so the edge gather is mostly L2-resident):
//   1. cnt[dst*R+r] edge counts (hw f32 atomics)
//   2. T1[r] = X @ W1[r] (bf16 WMMA, f32 acc); slot 8 = X@root1 + b1 (= agg buffer)
//   3. edge scatter: agg += T1[et][src] / max(cnt,1)   (hw global_atomic_add_f32)
//   4. T2[r] = relu(agg) @ W2[r]; slot 8 -> d_out = relu(agg)@root2 + b2
//   5. edge scatter into d_out
//
// Workspace: cnt (N*8) + T1 (9*N*256) + T2 (8*N*128) floats ~= 267 MB.

#define IN_C  256
#define HID_C 256
#define OUT_CH 128
#define NREL  8

typedef __attribute__((ext_vector_type(16))) __bf16 v16bf;
typedef __attribute__((ext_vector_type(8)))  __bf16 v8bf;
typedef __attribute__((ext_vector_type(8)))  float  v8f;

// Guaranteed-hardware float atomic add (no-return form, device scope).
// Avoids the atomicrmw->CAS-loop expansion risk on the bottleneck phase.
__device__ __forceinline__ void gatomic_fadd(float* p, float v) {
    asm volatile("global_atomic_add_f32 %0, %1, off scope:SCOPE_DEV"
                 :: "v"(p), "v"(v) : "memory");
}

// ---------------------------------------------------------------- utilities
__global__ __launch_bounds__(256) void rgcn_zero(float* __restrict__ p, int n) {
    int i = blockIdx.x * 256 + threadIdx.x;
    if (i < n) p[i] = 0.0f;
}

__global__ __launch_bounds__(256) void rgcn_count(const int* __restrict__ ei,
                                                  const int* __restrict__ et,
                                                  float* __restrict__ cnt, int E) {
    int e = blockIdx.x * 256 + threadIdx.x;
    if (e >= E) return;
    int dst = ei[E + e];
    int r   = et[e];
    gatomic_fadd(&cnt[dst * NREL + r], 1.0f);
}

// -------------------------------------------------- WMMA per-relation GEMM
// Block: 256 threads (8 wave32). blockIdx.x: 128-row node tile, blockIdx.y:
// slot 0..8 (0..7 = relations, 8 = root+bias). Wave w computes rows
// [tile+16w, tile+16w+16) x all OUTC columns with v_wmma_f32_16x16x32_bf16.
template<int INC, int OUTC, bool RELU>
__global__ __launch_bounds__(256)
void rgcn_transform(const float* __restrict__ A,     // [N][INC]
                    const float* __restrict__ W,     // [R][INC][OUTC]
                    const float* __restrict__ root,  // [INC][OUTC]
                    const float* __restrict__ bias,  // [OUTC]
                    float* __restrict__ outT,        // [R][N][OUTC]
                    float* __restrict__ outRoot,     // [N][OUTC]
                    int nRows) {
    constexpr int KC    = 64;       // K chunk staged in LDS
    constexpr int PITCH = KC + 8;   // bf16 row pitch: 144B rows, 16B aligned
    __shared__ __bf16 ldsW[OUTC * PITCH];

    const int slot = blockIdx.y;
    const float* Wm = (slot < NREL) ? (W + (size_t)slot * INC * OUTC) : root;
    float* outBase  = (slot < NREL) ? (outT + (size_t)slot * (size_t)nRows * OUTC)
                                    : outRoot;

    const int mBase  = blockIdx.x * 128;
    const int tid    = threadIdx.x;
    const int wave   = tid >> 5;
    const int lane   = tid & 31;
    const int half   = lane >> 4;     // 0: lanes 0-15, 1: lanes 16-31
    const int lane16 = lane & 15;

    constexpr int NT = OUTC / 16;     // 16-column output tiles per wave
    v8f acc[NT] = {};

    const int  rowA  = mBase + wave * 16 + lane16;
    const bool rowOK = rowA < nRows;

    for (int kc = 0; kc < INC; kc += KC) {
        __syncthreads();   // previous chunk's readers done
        // ---- stage W chunk to LDS, transposed [n][k], f32 -> bf16 ----
        {
            constexpr int NSL = 256 / OUTC;          // 1 (OUTC=256) or 2 (128)
            const int n  = tid % OUTC;
            const int ks = (tid / OUTC) * (KC / NSL);
            #pragma unroll
            for (int kk = 0; kk < KC / NSL; ++kk) {
                int k = ks + kk;
                ldsW[n * PITCH + k] = (__bf16)Wm[(size_t)(kc + k) * OUTC + n];
            }
        }
        __syncthreads();

        #pragma unroll
        for (int k0 = 0; k0 < KC; k0 += 32) {
            // ---- A fragment: 16x32 bf16 per ISA layout ----
            // lanes 0-15: K {0..7, 16..23}; lanes 16-31: K {8..15, 24..31}
            float p[8], q[8];
            if (rowOK) {
                const float* rp = A + (size_t)rowA * INC + (kc + k0 + half * 8);
                float4 x0 = *(const float4*)(rp + 0);
                float4 x1 = *(const float4*)(rp + 4);
                float4 y0 = *(const float4*)(rp + 16);
                float4 y1 = *(const float4*)(rp + 20);
                p[0]=x0.x; p[1]=x0.y; p[2]=x0.z; p[3]=x0.w;
                p[4]=x1.x; p[5]=x1.y; p[6]=x1.z; p[7]=x1.w;
                q[0]=y0.x; q[1]=y0.y; q[2]=y0.z; q[3]=y0.w;
                q[4]=y1.x; q[5]=y1.y; q[6]=y1.z; q[7]=y1.w;
            } else {
                #pragma unroll
                for (int i = 0; i < 8; ++i) { p[i] = 0.0f; q[i] = 0.0f; }
            }
            if (RELU) {
                #pragma unroll
                for (int i = 0; i < 8; ++i) {
                    p[i] = fmaxf(p[i], 0.0f);
                    q[i] = fmaxf(q[i], 0.0f);
                }
            }
            v16bf a;
            #pragma unroll
            for (int i = 0; i < 8; ++i) {
                a[i]     = (__bf16)p[i];
                a[i + 8] = (__bf16)q[i];
            }

            // ---- B fragments from LDS (col-major bf16) + WMMA ----
            const int kB = k0 + half * 8;
            #pragma unroll
            for (int j = 0; j < NT; ++j) {
                const int col = j * 16 + lane16;
                v8bf lo = *(const v8bf*)&ldsW[col * PITCH + kB];
                v8bf hi = *(const v8bf*)&ldsW[col * PITCH + kB + 16];
                v16bf b = __builtin_shufflevector(lo, hi,
                          0,1,2,3,4,5,6,7,8,9,10,11,12,13,14,15);
                acc[j] = __builtin_amdgcn_wmma_f32_16x16x32_bf16(
                             false, a, false, b, (short)0, acc[j], false, false);
            }
        }
    }

    // ---- store D (16x16 f32 C/D layout: VGPR v -> row v / v+8) ----
    const int rowD0 = mBase + wave * 16 + half * 8;
    #pragma unroll
    for (int j = 0; j < NT; ++j) {
        const int col = j * 16 + lane16;
        const float bv = (slot < NREL) ? 0.0f : bias[col];
        #pragma unroll
        for (int v = 0; v < 8; ++v) {
            int row = rowD0 + v;
            if (row < nRows)
                outBase[(size_t)row * OUTC + col] = acc[j][v] + bv;
        }
    }
}

// -------------------------------------------- edge gather + mean scatter-add
// One wave32 per edge; lanes cover the channel dimension with float4 reads,
// scatter with hardware global_atomic_add_f32 (no-return, device scope).
template<int CH>
__global__ __launch_bounds__(256)
void rgcn_edge_agg(const float* __restrict__ T,    // [R][N][CH]
                   const float* __restrict__ cnt,  // [N][R]
                   const int* __restrict__ ei,     // [2][E]
                   const int* __restrict__ et,     // [E]
                   float* __restrict__ out,        // [N][CH]
                   int nRows, int E) {
    int e    = blockIdx.x * 8 + (threadIdx.x >> 5);
    int lane = threadIdx.x & 31;
    if (e >= E) return;
    int src = ei[e];
    int dst = ei[E + e];
    int r   = et[e];
    float c    = cnt[dst * NREL + r];
    float norm = 1.0f / fmaxf(c, 1.0f);
    const float* msg = T + ((size_t)r * nRows + src) * CH;
    float* o = out + (size_t)dst * CH;
    #pragma unroll
    for (int g = 0; g < CH / 128; ++g) {
        int ch = g * 128 + lane * 4;
        float4 m = *(const float4*)(msg + ch);
        gatomic_fadd(o + ch + 0, m.x * norm);
        gatomic_fadd(o + ch + 1, m.y * norm);
        gatomic_fadd(o + ch + 2, m.z * norm);
        gatomic_fadd(o + ch + 3, m.w * norm);
    }
}

// ------------------------------------------------------------------ launch
extern "C" void kernel_launch(void* const* d_in, const int* in_sizes, int n_in,
                              void* d_out, int out_size, void* d_ws, size_t ws_size,
                              hipStream_t stream) {
    const float* x     = (const float*)d_in[0];
    const float* W1    = (const float*)d_in[1];
    const float* root1 = (const float*)d_in[2];
    const float* b1    = (const float*)d_in[3];
    const float* W2    = (const float*)d_in[4];
    const float* root2 = (const float*)d_in[5];
    const float* b2    = (const float*)d_in[6];
    const int*   ei    = (const int*)d_in[7];
    const int*   et    = (const int*)d_in[8];

    const int N = in_sizes[0] / IN_C;
    const int E = in_sizes[8];

    float* ws   = (float*)d_ws;
    float* cnt  = ws;                                   // N*8
    float* T1   = cnt + (size_t)N * NREL;               // 9*N*256 (slot 8 = agg)
    float* agg1 = T1 + (size_t)NREL * N * HID_C;        // alias: slot 8
    float* T2   = T1 + (size_t)(NREL + 1) * N * HID_C;  // 8*N*128
    float* out  = (float*)d_out;                        // slot-8 of layer 2

    const int nc = N * NREL;
    rgcn_zero<<<(nc + 255) / 256, 256, 0, stream>>>(cnt, nc);
    rgcn_count<<<(E + 255) / 256, 256, 0, stream>>>(ei, et, cnt, E);

    dim3 gT((N + 127) / 128, NREL + 1);
    rgcn_transform<IN_C, HID_C, false><<<gT, 256, 0, stream>>>(
        x, W1, root1, b1, T1, agg1, N);
    rgcn_edge_agg<HID_C><<<(E + 7) / 8, 256, 0, stream>>>(
        T1, cnt, ei, et, agg1, N, E);

    rgcn_transform<HID_C, OUT_CH, true><<<gT, 256, 0, stream>>>(
        agg1, W2, root2, b2, T2, out, N);
    rgcn_edge_agg<OUT_CH><<<(E + 7) / 8, 256, 0, stream>>>(
        T2, cnt, ei, et, out, N, E);
}
